// LSTMParser_35897336660783
// MI455X (gfx1250) — matrix operand order repaired and checked by reading, not compiled, not run on hardware
//
#include <hip/hip_runtime.h>

// ---------------------------------------------------------------------------
// LSTM (I=1024, H=1024, O=512, T=8192), batch=1, fp32 in/out.
// Plan:
//   K0: convert x, W_ih, W_hh, W_out to bf16 in ws; bsum = b_ih + b_hh
//   K1: px[T,4H]  = x_bf16 @ W_ih_bf16^T + bsum        (WMMA bf16 -> f32)
//   K2: persistent recurrence, 64 WGs, W_hh slice in LDS (128KB/WGP of 320KB),
//       grid barrier per step, emits h history as bf16
//   K3: logits[T,O] = h_bf16 @ W_out_bf16^T + b_out     (WMMA bf16 -> f32)
//   K4: log_softmax over axis=0 (per-column over T)
// ---------------------------------------------------------------------------

#define T_SEQ 8192
#define I_DIM 1024
#define H_DIM 1024
#define G_DIM 4096   // 4*H
#define O_DIM 512

// workspace layout (bytes, 256-aligned)
#define OFF_PX      0u                       // T*4H f32   = 134217728
#define OFF_HS      134217728u               // T*H  bf16  = 16777216
#define OFF_XBF     150994944u               // T*I  bf16  = 16777216
#define OFF_WIH     167772160u               // 4H*I bf16  = 8388608
#define OFF_WHH     176160768u               // 4H*H bf16  = 8388608
#define OFF_WOUT    184549376u               // O*H  bf16  = 1048576
#define OFF_BSUM    185597952u               // 4H f32     = 16384
#define OFF_LOGITS  185614336u               // T*O f32    = 16777216
#define OFF_HBUF    202391552u               // 2*H f32    = 8192
#define OFF_CNT     202399744u               // barrier counter

typedef __attribute__((ext_vector_type(16))) __bf16 v16bf;
typedef __attribute__((ext_vector_type(8)))  __bf16 v8bf;
typedef __attribute__((ext_vector_type(8)))  float  v8f;

__device__ __forceinline__ unsigned short f32_to_bf16(float f) {
  unsigned u = __float_as_uint(f);
  unsigned r = 0x7fffu + ((u >> 16) & 1u);   // round-to-nearest-even
  return (unsigned short)((u + r) >> 16);
}

// ------------------------- K0: conversions ---------------------------------
__global__ __launch_bounds__(256) void k_cvt_bf16_x4(const float* __restrict__ src,
                                                     unsigned short* __restrict__ dst,
                                                     int n4) {
  int i = blockIdx.x * 256 + threadIdx.x;
  if (i < n4) {
    float4 v = ((const float4*)src)[i];
    ushort4 o;
    o.x = f32_to_bf16(v.x); o.y = f32_to_bf16(v.y);
    o.z = f32_to_bf16(v.z); o.w = f32_to_bf16(v.w);
    ((ushort4*)dst)[i] = o;
  }
}

__global__ __launch_bounds__(256) void k_bias_sum(const float* __restrict__ a,
                                                  const float* __restrict__ b,
                                                  float* __restrict__ o, int n) {
  int i = blockIdx.x * 256 + threadIdx.x;
  if (i < n) o[i] = a[i] + b[i];
}

__global__ __launch_bounds__(256) void k_init_state(float* hbuf, unsigned* cnt) {
  int t = threadIdx.x;
  for (int i = t; i < 2 * H_DIM; i += 256) hbuf[i] = 0.0f;
  if (t < 64) cnt[t] = 0u;
}

// ------------------------- WMMA NT GEMM ------------------------------------
// C[M,N] = A[M,K] @ B[N,K]^T + bias[N], bf16 inputs, f32 out. K,M,N % 32 == 0.
// Block: 256 threads = 8 waves in 4(M) x 2(N); each wave computes 32x32
// (2x2 WMMA 16x16 tiles). Block tile = 128 x 64.

__device__ __forceinline__ v16bf load_a_frag(const __bf16* __restrict__ A,
                                             int lda, int m0, int kt) {
  const int lane  = threadIdx.x & 31;
  const int m     = m0 + (lane & 15);
  const int base0 = (lane < 16) ? 0 : 8;   // ISA 16-bit A 16x32 lane layout
  const __bf16* p = A + (size_t)m * lda + kt + base0;
  v8bf lo = *(const v8bf*)(p);             // K = base0 .. base0+7
  v8bf hi = *(const v8bf*)(p + 16);        // K = base0+16 .. base0+23
  v16bf r;
#pragma unroll
  for (int i = 0; i < 8; ++i) { r[i] = lo[i]; r[i + 8] = hi[i]; }
  return r;
}

__device__ __forceinline__ v16bf load_b_frag(const __bf16* __restrict__ W,
                                             int ldw, int n0, int kt) {
  const int lane = threadIdx.x & 31;
  const int n    = n0 + (lane & 15);
  const int koff = (lane < 16) ? 0 : 8;    // ISA 16-bit B KxN lane layout
  const __bf16* q = W + (size_t)n * ldw + kt + koff;
  v8bf lo = *(const v8bf*)(q);             // K = koff .. koff+7
  v8bf hi = *(const v8bf*)(q + 16);        // K = koff+16 .. koff+23
  v16bf r;
#pragma unroll
  for (int i = 0; i < 8; ++i) { r[i] = lo[i]; r[i + 8] = hi[i]; }
  return r;
}

__global__ __launch_bounds__(256) void k_gemm_nt_bf16(const __bf16* __restrict__ A,
                                                      const __bf16* __restrict__ B,
                                                      const float* __restrict__ bias,
                                                      float* __restrict__ C,
                                                      int M, int N, int K) {
  const int w  = threadIdx.x >> 5;
  const int wm = w & 3, wn = w >> 2;
  const int M0 = blockIdx.x * 128 + wm * 32;
  const int N0 = blockIdx.y * 64  + wn * 32;

  v8f acc[2][2] = {};
  for (int kt = 0; kt < K; kt += 32) {
    v16bf a0 = load_a_frag(A, K, M0,      kt);
    v16bf a1 = load_a_frag(A, K, M0 + 16, kt);
    v16bf b0 = load_b_frag(B, K, N0,      kt);
    v16bf b1 = load_b_frag(B, K, N0 + 16, kt);
    acc[0][0] = __builtin_amdgcn_wmma_f32_16x16x32_bf16(false, a0, false, b0,
                                                        (short)0, acc[0][0], false, false);
    acc[0][1] = __builtin_amdgcn_wmma_f32_16x16x32_bf16(false, a0, false, b1,
                                                        (short)0, acc[0][1], false, false);
    acc[1][0] = __builtin_amdgcn_wmma_f32_16x16x32_bf16(false, a1, false, b0,
                                                        (short)0, acc[1][0], false, false);
    acc[1][1] = __builtin_amdgcn_wmma_f32_16x16x32_bf16(false, a1, false, b1,
                                                        (short)0, acc[1][1], false, false);
  }

  // C/D layout: lanes 0-15 -> N=lane, M=v; lanes 16-31 -> N=lane-16, M=v+8
  const int lane = threadIdx.x & 31;
  const int ncol = lane & 15;
  const int mhi  = (lane >> 4) * 8;
#pragma unroll
  for (int mi = 0; mi < 2; ++mi)
#pragma unroll
    for (int ni = 0; ni < 2; ++ni) {
      const int nn = N0 + ni * 16 + ncol;
      const float bv = bias[nn];
#pragma unroll
      for (int v = 0; v < 8; ++v) {
        const int mm = M0 + mi * 16 + mhi + v;
        C[(size_t)mm * N + nn] = acc[mi][ni][v] + bv;
      }
    }
}

// ------------------------- K2: persistent recurrence -----------------------
#define RNN_NWG 64
#define RNN_THR 256
// dyn LDS: wlds u16[64*1024] | hlds f32[1024] | glds f32[64] | clds f32[16]
#define RNN_LDS_BYTES (64 * 1024 * 2 + 1024 * 4 + 64 * 4 + 16 * 4)

__global__ __launch_bounds__(RNN_THR) void k_lstm_recurrent(
    const float* __restrict__ px, const unsigned short* __restrict__ Whh,
    float* __restrict__ hbuf, unsigned short* __restrict__ hs,
    unsigned* __restrict__ cnt) {
  extern __shared__ char smem[];
  unsigned short* wlds = (unsigned short*)smem;              // 131072 B
  float* hlds = (float*)(smem + 64 * 1024 * 2);              // 4096 B
  float* glds = hlds + 1024;                                 // 256 B
  float* clds = glds + 64;                                   // 64 B

  const int tid = threadIdx.x;
  const int j0  = blockIdx.x * 16;       // this WG owns h[j0 .. j0+15]
  const int r   = tid >> 2;              // LDS row 0..63 (gate = r>>4, jj = r&15)
  const int p   = tid & 3;               // K partition (256 each)

  // one-time: stage 64 rows of W_hh (rows gate*1024 + j0 + jj) into LDS
  {
    const int grow = (r >> 4) * 1024 + j0 + (r & 15);
    const unsigned short* src = Whh + (size_t)grow * 1024 + p * 256;
    unsigned short* dst = wlds + r * 1024 + p * 256;
#pragma unroll 4
    for (int k = 0; k < 256; k += 8)
      *(uint4*)(dst + k) = *(const uint4*)(src + k);
  }
  if (tid < 16) clds[tid] = 0.0f;
  __syncthreads();

  const unsigned short* wrow = wlds + r * 1024 + p * 256;
  const float* hp = hlds + p * 256;

  for (int t = 0; t < T_SEQ; ++t) {
    // stage h(t) into LDS (4 floats / thread)
    *(float4*)(hlds + tid * 4) = *(const float4*)(hbuf + (t & 1) * H_DIM + tid * 4);
    __syncthreads();

    // partial dot: row r over K in [p*256, p*256+256)
    float acc = 0.0f;
#pragma unroll 8
    for (int k = 0; k < 256; k += 4) {
      uint2 w2 = *(const uint2*)(wrow + k);          // 4 packed bf16
      float4 hv = *(const float4*)(hp + k);
      acc += __uint_as_float(w2.x << 16)          * hv.x;
      acc += __uint_as_float(w2.x & 0xffff0000u)  * hv.y;
      acc += __uint_as_float(w2.y << 16)          * hv.z;
      acc += __uint_as_float(w2.y & 0xffff0000u)  * hv.w;
    }
    acc += __shfl_xor(acc, 1, 32);
    acc += __shfl_xor(acc, 2, 32);
    if (p == 0) glds[r] = acc;
    __syncthreads();

    if (tid < 16) {
      const int j = j0 + tid;
      const float* pxt = px + (size_t)t * G_DIM;
      float gi = glds[tid]      + pxt[j];
      float gf = glds[16 + tid] + pxt[1024 + j];
      float gg = glds[32 + tid] + pxt[2048 + j];
      float go = glds[48 + tid] + pxt[3072 + j];
      float i_ = 1.0f / (1.0f + __expf(-gi));
      float f_ = 1.0f / (1.0f + __expf(-gf));
      float g_ = tanhf(gg);
      float o_ = 1.0f / (1.0f + __expf(-go));
      float c  = f_ * clds[tid] + i_ * g_;
      clds[tid] = c;
      float h  = o_ * tanhf(c);
      hbuf[((t + 1) & 1) * H_DIM + j] = h;                 // h(t+1) buffer
      hs[(size_t)t * H_DIM + j] = f32_to_bf16(h);          // history for GEMM2
    }
    __syncthreads();

    // grid-wide barrier: monotonic counter, target = (t+1)*NWG
    if (tid == 0) {
      __threadfence();
      atomicAdd(cnt, 1u);
      const unsigned target = (unsigned)(t + 1) * RNN_NWG;
      while (__hip_atomic_load(cnt, __ATOMIC_ACQUIRE, __HIP_MEMORY_SCOPE_AGENT) < target)
        __builtin_amdgcn_s_sleep(2);
    }
    __syncthreads();
  }
}

// ------------------------- K4: log_softmax over axis=0 ---------------------
__global__ __launch_bounds__(256) void k_logsoftmax_col(const float* __restrict__ logits,
                                                        float* __restrict__ out) {
  __shared__ float red[256];
  const int col = blockIdx.x, tid = threadIdx.x;

  float m = -3.4e38f;
  for (int t = tid; t < T_SEQ; t += 256)
    m = fmaxf(m, logits[(size_t)t * O_DIM + col]);
  red[tid] = m; __syncthreads();
  for (int s = 128; s > 0; s >>= 1) {
    if (tid < s) red[tid] = fmaxf(red[tid], red[tid + s]);
    __syncthreads();
  }
  m = red[0]; __syncthreads();

  float sum = 0.0f;
  for (int t = tid; t < T_SEQ; t += 256)
    sum += __expf(logits[(size_t)t * O_DIM + col] - m);
  red[tid] = sum; __syncthreads();
  for (int s = 128; s > 0; s >>= 1) {
    if (tid < s) red[tid] += red[tid + s];
    __syncthreads();
  }
  const float lse = m + __logf(red[0]);
  __syncthreads();

  for (int t = tid; t < T_SEQ; t += 256) {
    const size_t idx = (size_t)t * O_DIM + col;
    out[idx] = logits[idx] - lse;
  }
}

// ---------------------------------------------------------------------------
extern "C" void kernel_launch(void* const* d_in, const int* in_sizes, int n_in,
                              void* d_out, int out_size, void* d_ws, size_t ws_size,
                              hipStream_t stream) {
  const float* x     = (const float*)d_in[0];
  const float* W_ih  = (const float*)d_in[1];
  const float* W_hh  = (const float*)d_in[2];
  const float* b_ih  = (const float*)d_in[3];
  const float* b_hh  = (const float*)d_in[4];
  const float* W_out = (const float*)d_in[5];
  const float* b_out = (const float*)d_in[6];

  char* ws = (char*)d_ws;
  float*          px     = (float*)(ws + OFF_PX);
  unsigned short* hs     = (unsigned short*)(ws + OFF_HS);
  unsigned short* xbf    = (unsigned short*)(ws + OFF_XBF);
  unsigned short* wih    = (unsigned short*)(ws + OFF_WIH);
  unsigned short* whh    = (unsigned short*)(ws + OFF_WHH);
  unsigned short* wout   = (unsigned short*)(ws + OFF_WOUT);
  float*          bsum   = (float*)(ws + OFF_BSUM);
  float*          logits = (float*)(ws + OFF_LOGITS);
  float*          hbuf   = (float*)(ws + OFF_HBUF);
  unsigned*       cnt    = (unsigned*)(ws + OFF_CNT);

  // K0: conversions
  {
    int n4;
    n4 = (T_SEQ * I_DIM) / 4;
    k_cvt_bf16_x4<<<(n4 + 255) / 256, 256, 0, stream>>>(x, xbf, n4);
    n4 = (G_DIM * I_DIM) / 4;
    k_cvt_bf16_x4<<<(n4 + 255) / 256, 256, 0, stream>>>(W_ih, wih, n4);
    n4 = (G_DIM * H_DIM) / 4;
    k_cvt_bf16_x4<<<(n4 + 255) / 256, 256, 0, stream>>>(W_hh, whh, n4);
    n4 = (O_DIM * H_DIM) / 4;
    k_cvt_bf16_x4<<<(n4 + 255) / 256, 256, 0, stream>>>(W_out, wout, n4);
    k_bias_sum<<<G_DIM / 256, 256, 0, stream>>>(b_ih, b_hh, bsum, G_DIM);
    k_init_state<<<1, 256, 0, stream>>>(hbuf, cnt);
  }

  // K1: px = x @ W_ih^T + (b_ih + b_hh)
  k_gemm_nt_bf16<<<dim3(T_SEQ / 128, G_DIM / 64), 256, 0, stream>>>(
      (const __bf16*)xbf, (const __bf16*)wih, bsum, px, T_SEQ, G_DIM, I_DIM);

  // K2: sequential LSTM (persistent, W_hh slice LDS-resident)
  k_lstm_recurrent<<<RNN_NWG, RNN_THR, RNN_LDS_BYTES, stream>>>(px, whh, hbuf, hs, cnt);

  // K3: logits = h @ W_out^T + b_out
  k_gemm_nt_bf16<<<dim3(T_SEQ / 128, O_DIM / 64), 256, 0, stream>>>(
      (const __bf16*)hs, (const __bf16*)wout, b_out, logits, T_SEQ, O_DIM, H_DIM);

  // K4: log_softmax over sequence axis
  k_logsoftmax_col<<<O_DIM, 256, 0, stream>>>(logits, (float*)d_out);
}
